// GlobalModel_18786186952966
// MI455X (gfx1250) — compile-verified, matrix operand order checked by reading, not run on hardware
//
#include <hip/hip_runtime.h>
#include <math.h>

// CDNA5 / gfx1250: wave32, WMMA 16x16x32 bf16 -> f32 accumulate.
typedef __attribute__((ext_vector_type(16))) __bf16 v16bf;
typedef __attribute__((ext_vector_type(8)))  float  v8f;

#define D   128
#define H   8
#define DH  16
#define BN_EPS   1e-5f
#define ATTN_EPS 1e-6f

// Packed weight fragments: [mat(4)][ntile(8)][ks(4)][lane(32)][16 bf16]
#define WB_FRAG_ELEMS   (32 * 16)
#define WB_MAT_ELEMS    (8 * 4 * WB_FRAG_ELEMS)
#define WB_TOTAL_ELEMS  (4 * WB_MAT_ELEMS)              // 65536 bf16 = 128 KB

// ---------------------------------------------------------------------------
// WMMA fragment layouts per CDNA5 ISA 7.12.2 (wave32):
//  A (16x32, 16-bit): lane&15 = row M; lane>>4 selects K halves {0-7,16-23}
//                     vs {8-15,24-31}; v16bf elems 0-7 = K kb..kb+7,
//                     elems 8-15 = K kb+16..kb+23.
//  B (32x16, 16-bit): mirrored layout, lane&15 = column N.
//  C/D (16x16 f32):   VGPR r, lanes 0-15 -> (M=r, N=lane); lanes 16-31 ->
//                     (M=r+8, N=lane-16).
// ---------------------------------------------------------------------------

// Kernel P: pack one weight fragment per 32-thread block into bf16
// B-fragment order.  unit = mat*32 + ntile*4 + ks  (128 blocks total).
__global__ void pack_w_kernel(const float* __restrict__ WQ,
                              const float* __restrict__ WK,
                              const float* __restrict__ WV,
                              const float* __restrict__ WE,
                              __bf16* __restrict__ Wb) {
    int unit  = blockIdx.x;            // 0..127
    int mat   = unit >> 5;
    int ntile = (unit >> 2) & 7;
    int ks    = unit & 3;
    const float* W = (mat == 0) ? WQ : (mat == 1) ? WK : (mat == 2) ? WV : WE;
    int lane = threadIdx.x;            // 0..31
    int col  = ntile * 16 + (lane & 15);
    int kb   = ks * 32 + ((lane >> 4) << 3);
    __bf16* dst = Wb + ((size_t)unit * 32 + lane) * 16;
#pragma unroll
    for (int j = 0; j < 8; ++j) {
        dst[j]     = (__bf16)W[(size_t)(kb + j) * D + col];
        dst[8 + j] = (__bf16)W[(size_t)(kb + 16 + j) * D + col];
    }
}

__device__ __forceinline__ const v16bf* wb_frag(const __bf16* Wb, int mat,
                                                int ntile, int ks) {
    return (const v16bf*)(Wb + ((size_t)((mat * 8 + ntile) * 4 + ks)) * WB_FRAG_ELEMS);
}

// Cooperative A-tile staging: 256 threads convert a 16x128 fp32 tile into
// bf16 A-fragment order in LDS: lA[ks][lane][16].  Thread t fills 8 elems.
__device__ __forceinline__ void stage_a_tile(const float* __restrict__ A,
                                             int rowBase, int maxRow,
                                             __bf16 (*lA)[32][16], int t) {
    int ks   = t >> 6;                 // 0..3
    int lane = (t >> 1) & 31;          // fragment lane
    int half = t & 1;                  // elems 0-7 vs 8-15
    int row = rowBase + (lane & 15);
    if (row >= maxRow) row = maxRow - 1;   // clamp; EXEC stays all-ones
    int k0 = ks * 32 + ((lane >> 4) << 3) + half * 16;
    const float* p = A + (size_t)row * D + k0;
    __bf16* q = &lA[ks][lane][half * 8];
#pragma unroll
    for (int j = 0; j < 8; ++j) q[j] = (__bf16)p[j];
}

// ---------------------------------------------------------------------------
// Kernel 0: zero the accumulation scratch (wV, Z, colsum, colsumsq).
// ---------------------------------------------------------------------------
__global__ void zero_kernel(float* __restrict__ p, size_t n) {
    size_t i = (size_t)blockIdx.x * blockDim.x + threadIdx.x;
    size_t stride = (size_t)gridDim.x * blockDim.x;
    for (; i < n; i += stride) p[i] = 0.0f;
}

// ---------------------------------------------------------------------------
// Kernel 1: Q,K,V = x @ {WQ,WK,WV}.  Block = 256 threads = 8 waves; block
// owns a 16-row tile of x (staged once in LDS); wave w owns columns
// [16w,16w+16).  A fragments re-read from LDS per k-step (low reg pressure),
// B fragments are two coalesced b128 loads from the packed bf16 weights.
// ---------------------------------------------------------------------------
__global__ void proj_qkv_kernel(const float* __restrict__ x,
                                const __bf16* __restrict__ Wb,
                                float* __restrict__ Q,
                                float* __restrict__ K,
                                float* __restrict__ V,
                                int N) {
    __shared__ alignas(32) __bf16 lA[4][32][16];
    int t = threadIdx.x;
    int lane = t & 31;
    int wave = t >> 5;                 // output column tile
    int m0 = blockIdx.x * 16;

    stage_a_tile(x, m0, N, lA, t);
    __syncthreads();

    int hi8 = (lane >> 4) << 3;
    int col = wave * 16 + (lane & 15);
    bool full = (m0 + 16 <= N);        // uniform across block

    float* Os[3] = {Q, K, V};
#pragma unroll
    for (int m = 0; m < 3; ++m) {
        v8f c = {};
#pragma unroll
        for (int ks = 0; ks < 4; ++ks) {
            v16bf a = *(const v16bf*)&lA[ks][lane][0];
            v16bf b = wb_frag(Wb, m, wave, ks)[lane];
            c = __builtin_amdgcn_wmma_f32_16x16x32_bf16(
                    false, a, false, b, (short)0, c, false, false);
        }
        float* out = Os[m];
        if (full) {
#pragma unroll
            for (int r = 0; r < 8; ++r)
                out[(size_t)(m0 + r + hi8) * D + col] = c[r];
        } else {
#pragma unroll
            for (int r = 0; r < 8; ++r) {
                int row = m0 + r + hi8;
                if (row < N) out[(size_t)row * D + col] = c[r];
            }
        }
    }
}

// ---------------------------------------------------------------------------
// Kernel 2: fused edge stage.  Block = 16 edges, 8 waves; wave w == head w
// (dh=16 == the WMMA N-tile).  Eh tile computed on the fly via 4 WMMAs and
// never materialized (saves ~600 MB of HBM traffic vs a separate Eh GEMM).
// score = exp(clip(sum_dh K[src]*Q[dst]*Eh/4, -5, 5)) via 16-lane xor tree;
// msg = V[src]*score scatter-added to wV[dst] (25.6 MB -> L2-resident).
// ---------------------------------------------------------------------------
__global__ void edge_attn_kernel(const float* __restrict__ edge_attr,
                                 const __bf16* __restrict__ Wb,
                                 const float* __restrict__ Q,
                                 const float* __restrict__ K,
                                 const float* __restrict__ V,
                                 const int* __restrict__ src,
                                 const int* __restrict__ dst,
                                 float* __restrict__ wV,
                                 float* __restrict__ Zacc,
                                 int E) {
    __shared__ alignas(32) __bf16 lA[4][32][16];
    int t = threadIdx.x;
    int lane = t & 31;
    int head = t >> 5;                 // wave id == head id
    int e0 = blockIdx.x * 16;

    stage_a_tile(edge_attr, e0, E, lA, t);
    __syncthreads();

    v8f c = {};
#pragma unroll
    for (int ks = 0; ks < 4; ++ks) {
        v16bf a = *(const v16bf*)&lA[ks][lane][0];
        v16bf b = wb_frag(Wb, 3, head, ks)[lane];   // WE fragments
        c = __builtin_amdgcn_wmma_f32_16x16x32_bf16(
                false, a, false, b, (short)0, c, false, false);
    }

    int hi8 = (lane >> 4) << 3;
    int col = head * DH + (lane & 15);
    bool full = (e0 + 16 <= E);        // uniform across block

#pragma unroll
    for (int r = 0; r < 8; ++r) {
        int e = e0 + r + hi8;
        int ec = (full || e < E) ? e : (E - 1);
        int s = src[ec];
        int d = dst[ec];
        // (K[src] * Q[dst]) / sqrt(dh) * Eh   (1/sqrt(16) = 0.25)
        float val = K[(size_t)s * D + col] * Q[(size_t)d * D + col]
                    * 0.25f * c[r];
        // reduce over this head's 16 columns (xor tree within half-waves)
#pragma unroll
        for (int off = 1; off < 16; off <<= 1)
            val += __shfl_xor(val, off, 32);
        float sc = __expf(fminf(fmaxf(val, -5.0f), 5.0f));
        if (full || e < E) {
            float msg = V[(size_t)s * D + col] * sc;
            atomicAdd(&wV[(size_t)d * D + col], msg);
            if ((lane & 15) == 0)
                atomicAdd(&Zacc[(size_t)d * H + head], sc);
        }
    }
}

// ---------------------------------------------------------------------------
// Kernel 3: h = x + wV/(Z+eps); per-column sum / sumsq partials.
// 128 threads = one per column (coalesced row sweeps); one atomic per column
// per block.
// ---------------------------------------------------------------------------
__global__ void h_colstats_kernel(const float* __restrict__ x,
                                  const float* __restrict__ wV,
                                  const float* __restrict__ Zacc,
                                  float* __restrict__ h,
                                  float* __restrict__ colsum,
                                  float* __restrict__ colsumsq,
                                  int N, int rowsPerBlock) {
    int c = threadIdx.x;               // 0..127
    int head = c >> 4;
    int r0 = blockIdx.x * rowsPerBlock;
    int r1 = r0 + rowsPerBlock;
    if (r1 > N) r1 = N;
    float s = 0.0f, sq = 0.0f;
    for (int r = r0; r < r1; ++r) {
        float z  = Zacc[(size_t)r * H + head];
        float hv = x[(size_t)r * D + c] + wV[(size_t)r * D + c] / (z + ATTN_EPS);
        h[(size_t)r * D + c] = hv;
        s += hv;
        sq += hv * hv;
    }
    atomicAdd(&colsum[c], s);
    atomicAdd(&colsumsq[c], sq);
}

// ---------------------------------------------------------------------------
// Kernel 4: BatchNorm (training stats): out = (h-mean)*rsqrt(var+eps)*g + b.
// var = E[h^2] - mean^2 (same as reference's mean((h-mean)^2)).
// ---------------------------------------------------------------------------
__global__ void bn_kernel(const float* __restrict__ h,
                          const float* __restrict__ colsum,
                          const float* __restrict__ colsumsq,
                          const float* __restrict__ gamma,
                          const float* __restrict__ beta,
                          float* __restrict__ out, int N) {
    size_t total = (size_t)N * D;
    size_t stride = (size_t)gridDim.x * blockDim.x;
    float invN = 1.0f / (float)N;
    for (size_t i = (size_t)blockIdx.x * blockDim.x + threadIdx.x;
         i < total; i += stride) {
        int c = (int)(i & (D - 1));
        float mean = colsum[c] * invN;
        float var  = colsumsq[c] * invN - mean * mean;
        out[i] = (h[i] - mean) * rsqrtf(var + BN_EPS) * gamma[c] + beta[c];
    }
}

// ---------------------------------------------------------------------------
extern "C" void kernel_launch(void* const* d_in, const int* in_sizes, int n_in,
                              void* d_out, int out_size, void* d_ws, size_t ws_size,
                              hipStream_t stream) {
    const float* x         = (const float*)d_in[0];
    const float* edge_attr = (const float*)d_in[1];
    const float* WQ        = (const float*)d_in[2];
    const float* WK        = (const float*)d_in[3];
    const float* WV        = (const float*)d_in[4];
    const float* WE        = (const float*)d_in[5];
    const float* gamma     = (const float*)d_in[6];
    const float* beta      = (const float*)d_in[7];
    const int*   edge_idx  = (const int*)d_in[8];

    const int N = in_sizes[0] / D;     // 50000
    const int E = in_sizes[1] / D;     // 600000
    const int* src = edge_idx;         // edge_index[0]
    const int* dst = edge_idx + E;     // edge_index[1]

    // Workspace layout: packed bf16 weights first, then float arrays.
    __bf16* Wb = (__bf16*)d_ws;                      // 128 KB
    float* ws = (float*)((char*)d_ws + (size_t)WB_TOTAL_ELEMS * sizeof(__bf16));
    size_t nd = (size_t)N * D;
    float* Q        = ws;                 ws += nd;
    float* K        = ws;                 ws += nd;
    float* V        = ws;                 ws += nd;
    float* h        = ws;                 ws += nd;
    float* wV       = ws;                 ws += nd;            // zeroed
    float* Zacc     = ws;                 ws += (size_t)N * H; // zeroed
    float* colsum   = ws;                 ws += D;             // zeroed
    float* colsumsq = ws;                 ws += D;             // zeroed
    size_t zeroCount = nd + (size_t)N * H + 2 * D;

    pack_w_kernel<<<128, 32, 0, stream>>>(WQ, WK, WV, WE, Wb);
    zero_kernel<<<1024, 256, 0, stream>>>(wV, zeroCount);

    proj_qkv_kernel<<<(N + 15) / 16, 256, 0, stream>>>(x, Wb, Q, K, V, N);

    edge_attn_kernel<<<(E + 15) / 16, 256, 0, stream>>>(
        edge_attr, Wb, Q, K, V, src, dst, wV, Zacc, E);

    const int rowsPerBlock = 256;
    h_colstats_kernel<<<(N + rowsPerBlock - 1) / rowsPerBlock, D, 0, stream>>>(
        x, wV, Zacc, h, colsum, colsumsq, N, rowsPerBlock);

    bn_kernel<<<2048, 256, 0, stream>>>(
        h, colsum, colsumsq, gamma, beta, (float*)d_out, N);
}
